// Interfusion_GRU_9079560863792
// MI455X (gfx1250) — matrix operand order repaired and checked by reading, not compiled, not run on hardware
//
#include <hip/hip_runtime.h>
#include <hip/hip_bf16.h>

typedef __bf16 bf16_t;
typedef __bf16 v16bf __attribute__((ext_vector_type(16)));
typedef __bf16 v8bf  __attribute__((ext_vector_type(8)));
typedef float  v8f   __attribute__((ext_vector_type(8)));
typedef float  v4f   __attribute__((ext_vector_type(4)));

#define GRU_D 64
#define LOG2E 1.44269504088896340736f

static __device__ __forceinline__ v8f wmma_bf16(v16bf a, v16bf b, v8f c) {
  // D = A(16x32 bf16) x B(32x16 bf16) + C(16x16 f32)
  return __builtin_amdgcn_wmma_f32_16x16x32_bf16(false, a, false, b,
                                                 (short)0, c, false, false);
}

// Fast device math: native v_exp_f32 / v_rcp_f32 (TRANS ops, co-execute).
static __device__ __forceinline__ float fexp2(float x) {
  return __builtin_amdgcn_exp2f(x);
}
static __device__ __forceinline__ float frcp(float x) {
  return __builtin_amdgcn_rcpf(x);
}

// One wave32 handles one 16-row block through all 3 GRU cells.
// 256 threads = 8 waves = 128 rows per workgroup.
__global__ __launch_bounds__(256) void gru3_fused(
    const float* __restrict__ x0, const float* __restrict__ x1,
    const float* __restrict__ x2, const float* __restrict__ w_ih,
    const float* __restrict__ w_hh, const float* __restrict__ b_ih,
    const float* __restrict__ b_hh, float* __restrict__ out, int nrows) {
  // Weights as bf16 in LDS: [2][192 rows][64 cols], row-major. 48 KB.
  __shared__ __align__(16) bf16_t sW[2][192 * 64];
  // Per-wave h transpose staging (C-layout -> A-layout), bf16. 16 KB.
  __shared__ __align__(16) bf16_t sH[8][16 * 64];

  const int tid = threadIdx.x;
  // Cooperative f32 -> bf16 weight staging.
  for (int idx = tid; idx < 192 * 64; idx += 256) {
    sW[0][idx] = (bf16_t)w_ih[idx];
    sW[1][idx] = (bf16_t)w_hh[idx];
  }
  __syncthreads();

  const int lane = tid & 31;
  const int wave = tid >> 5;
  const int lc = lane & 15;  // column within tile (B/C/D); row for A
  const int hi = lane >> 4;  // lane group 0/1
  const long rowBase = ((long)blockIdx.x * 8 + wave) * 16;
  if (rowBase >= nrows) return;  // after barrier: safe

  // Hint the streaming inputs for later cells into cache early.
  __builtin_prefetch(x1 + (rowBase + lc) * GRU_D, 0, 0);
  __builtin_prefetch(x2 + (rowBase + lc) * GRU_D, 0, 0);

  // Per-lane biases. r/z biases pre-scaled by -log2(e) so the sigmoid
  // exponent argument is a single fmaf.
  float bRs[4], bZs[4], bI[4], bH[4];
#pragma unroll
  for (int t = 0; t < 4; ++t) {
    bRs[t] = -(b_ih[t * 16 + lc] + b_hh[t * 16 + lc]) * LOG2E;
    bZs[t] = -(b_ih[64 + t * 16 + lc] + b_hh[64 + t * 16 + lc]) * LOG2E;
    bI[t] = b_ih[128 + t * 16 + lc];
    bH[t] = b_hh[128 + t * 16 + lc];
  }

  // A-fragment (16x32 bf16) from global f32, K-half kh.
  // lanes 0-15: K = kh*32 + {0..7,16..23}; lanes 16-31: +8 shift.
  auto loadA_global = [&](const float* base, int kh) -> v16bf {
    const float* p = base + (rowBase + lc) * GRU_D + kh * 32 + hi * 8;
    v4f f0 = *(const v4f*)(p);
    v4f f1 = *(const v4f*)(p + 4);
    v4f f2 = *(const v4f*)(p + 16);
    v4f f3 = *(const v4f*)(p + 20);
    v16bf a;
#pragma unroll
    for (int i = 0; i < 4; ++i) {
      a[i] = (bf16_t)f0[i];
      a[4 + i] = (bf16_t)f1[i];
      a[8 + i] = (bf16_t)f2[i];
      a[12 + i] = (bf16_t)f3[i];
    }
    return a;
  };

  // A-fragment of h from per-wave LDS staging (already bf16).
  // Union-based combine: register coalescing only, no per-element inserts.
  auto loadA_lds = [&](int kh) -> v16bf {
    const bf16_t* p = &sH[wave][0] + lc * GRU_D + kh * 32 + hi * 8;
    union {
      v16bf v;
      v8bf h[2];
    } u;
    u.h[0] = *(const v8bf*)(p);
    u.h[1] = *(const v8bf*)(p + 16);
    return u.v;
  };

  // B-fragment (32x16 bf16): lane holds weight row g*16+lc;
  // lanes 0-15: K = kh*32 + {0..7,16..23}; lanes 16-31: +8 shift.
  auto loadB = [&](int m, int g, int kh) -> v16bf {
    const bf16_t* p = &sW[m][0] + (g * 16 + lc) * GRU_D + kh * 32 + hi * 8;
    union {
      v16bf v;
      v8bf h[2];
    } u;
    u.h[0] = *(const v8bf*)(p);
    u.h[1] = *(const v8bf*)(p + 16);
    return u.v;
  };

  v8f hC[4];            // hidden state, C-layout f32 (tiles over d = 0..63)
  const v8f zero = {};  // folds to inline src2=0 in v_wmma

  auto cell = [&](const float* xsrc, bool hasH, bool storeH, bool storeOut) {
    v16bf a0 = loadA_global(xsrc, 0);
    v16bf a1 = loadA_global(xsrc, 1);
    v16bf h0{}, h1{};
    if (hasH) {
      h0 = loadA_lds(0);
      h1 = loadA_lds(1);
    }
    // One 16-column tile at a time: only 4 accumulators live at once.
    // Gate tiles in the 192-row weight matrix: r -> t, z -> t+4, n -> t+8.
#pragma unroll
    for (int t = 0; t < 4; ++t) {
      v8f accR = wmma_bf16(a0, loadB(0, t, 0), zero);
      accR = wmma_bf16(a1, loadB(0, t, 1), accR);
      v8f accZ = wmma_bf16(a0, loadB(0, t + 4, 0), zero);
      accZ = wmma_bf16(a1, loadB(0, t + 4, 1), accZ);
      v8f accI = wmma_bf16(a0, loadB(0, t + 8, 0), zero);
      accI = wmma_bf16(a1, loadB(0, t + 8, 1), accI);
      v8f accH = {};
      if (hasH) {
        accR = wmma_bf16(h0, loadB(1, t, 0), accR);
        accR = wmma_bf16(h1, loadB(1, t, 1), accR);
        accZ = wmma_bf16(h0, loadB(1, t + 4, 0), accZ);
        accZ = wmma_bf16(h1, loadB(1, t + 4, 1), accZ);
        accH = wmma_bf16(h0, loadB(1, t + 8, 0), zero);
        accH = wmma_bf16(h1, loadB(1, t + 8, 1), accH);
      }
      // Elementwise GRU epilogue for this tile.
#pragma unroll
      for (int e = 0; e < 8; ++e) {
        float r = frcp(1.0f + fexp2(fmaf(accR[e], -LOG2E, bRs[t])));
        float z = frcp(1.0f + fexp2(fmaf(accZ[e], -LOG2E, bZs[t])));
        float hn = hasH ? (accH[e] + bH[t]) : bH[t];
        float xn = fmaf(r, hn, accI[e] + bI[t]);
        float ex = fexp2(xn * (-2.0f * LOG2E));
        // tanh(x) = 2*sigmoid(2x) - 1 = 2/(1+ex) - 1
        float n = fmaf(2.0f, frcp(1.0f + ex), -1.0f);
        float hp = hasH ? hC[t][e] : 0.0f;
        float hnew = fmaf(z, hp - n, n);  // (1-z)*n + z*hp
        if (!storeOut) {
          hC[t][e] = hnew;
        }
        if (storeH) {
          sH[wave][(e + hi * 8) * GRU_D + t * 16 + lc] = (bf16_t)hnew;
        }
        if (storeOut) {
          out[(rowBase + e + hi * 8) * GRU_D + t * 16 + lc] = hnew;
        }
      }
      // Bound DS-prefetch distance to one tile: hold DS ops here, but let
      // VALU/SALU/WMMA/VMEM/TRANS schedule across (mask = 0x47F).
      __builtin_amdgcn_sched_barrier(0x47F);
    }
  };

  cell(x0, false, true, false);
  cell(x1, true, true, false);
  cell(x2, true, false, true);  // final cell streams h straight to out
}

extern "C" void kernel_launch(void* const* d_in, const int* in_sizes, int n_in,
                              void* d_out, int out_size, void* d_ws,
                              size_t ws_size, hipStream_t stream) {
  const float* x0 = (const float*)d_in[0];    // initial_node [B,A,64]
  const float* x1 = (const float*)d_in[1];    // internal_interaction
  const float* x2 = (const float*)d_in[2];    // external_interaction
  const float* w_ih = (const float*)d_in[3];  // [192,64]
  const float* w_hh = (const float*)d_in[4];  // [192,64]
  const float* b_ih = (const float*)d_in[5];  // [192]
  const float* b_hh = (const float*)d_in[6];  // [192]
  float* out = (float*)d_out;

  int nrows = in_sizes[0] / GRU_D;   // B*A
  int blocks = (nrows + 127) / 128;  // 128 rows per workgroup
  gru3_fused<<<blocks, 256, 0, stream>>>(x0, x1, x2, w_ih, w_hh, b_ih, b_hh,
                                         out, nrows);
}
